// BasisNetwork_81011673137403
// MI455X (gfx1250) — compile-verified, matrix operand order checked by reading, not run on hardware
//
#include <hip/hip_runtime.h>

typedef __attribute__((ext_vector_type(16))) _Float16 v16h;
typedef __attribute__((ext_vector_type(2)))  _Float16 h2;
typedef __attribute__((ext_vector_type(8)))  float    v8f;

#define N_NODES_C  100000
#define N_EDGES_C  800000
#define OUT_SCALE  (1.0f / 128.0f)

__global__ __launch_bounds__(256) void zero_f32_kernel(float* __restrict__ p, int n) {
    int i = blockIdx.x * blockDim.x + threadIdx.x;
    if (i < n) p[i] = 0.0f;
}

union AFrag {
    v16h v;
    h2   p[8];
};

// One wave32 processes a tile of 16 edges:
//   A (16x256, f16)  = per-edge basis-outer-product features, built in registers
//                      with packed f16 muls (x converted to f16 once per tile)
//   B (256x16, f16)  = Wflat ([K,F_in,F_out] viewed as [256,16] row-major),
//                      preloaded once per wave into 64 VGPRs in WMMA B layout
//   C (16x16,  f32)  = per-edge messages, 8 x v_wmma_f32_16x16x32_f16
__global__ __launch_bounds__(256) void basis_conv_kernel(
    const float* __restrict__ x,          // [N_NODES, 16]
    const float* __restrict__ edge_attr,  // [N_EDGES, 2]
    const float* __restrict__ W,          // [256, 16] row-major
    const int*   __restrict__ ei,         // [N_EDGES] dest
    const int*   __restrict__ ej,         // [N_EDGES] src
    float*       __restrict__ out)        // [N_NODES, 16]
{
    const int lane        = threadIdx.x & 31;
    const int waveInBlock = threadIdx.x >> 5;
    const int wavesPerBlk = blockDim.x >> 5;
    const int waveId      = blockIdx.x * wavesPerBlk + waveInBlock;
    const int numWaves    = gridDim.x * wavesPerBlk;

    const int half = lane >> 4;   // K-group selector within fragments
    const int ln   = lane & 15;   // A: edge row M   /  B,C: output column N

    // ---- Preload B fragments once, resident in VGPRs for the whole kernel ----
    // B layout (mirror of documented 16-bit A layout): lane holds column N = ln,
    // VGPR v holds K pair at K = s*32 + (v/4)*16 + half*8 + (v%4)*2.
    v16h Bfrag[8];
#pragma unroll
    for (int s = 0; s < 8; ++s) {
#pragma unroll
        for (int v = 0; v < 8; ++v) {
            const int g = v >> 2, w = v & 3;
            const int kLo = s * 32 + g * 16 + half * 8 + w * 2;
            Bfrag[s][2 * v]     = (_Float16)W[kLo * 16 + ln];
            Bfrag[s][2 * v + 1] = (_Float16)W[(kLo + 1) * 16 + ln];
        }
    }

    const int nTiles = N_EDGES_C / 16;   // exact: 50000 tiles
    for (int tile = waveId; tile < nTiles; tile += numWaves) {
        const int eBase = tile * 16;

        // --- edge owned by this lane for the A fragment: row M = ln ---
        const int eA = eBase + ln;
        const int jn = ej[eA];

        // hat basis on clamped coords: centers linspace(-1,1,4), width 2/3
        const float2 uv = *(const float2*)(edge_attr + eA * 2);
        const float u0 = fminf(1.0f, fmaxf(-1.0f, uv.x));
        const float u1 = fminf(1.0f, fmaxf(-1.0f, uv.y));
        float bx[4], by[4];
#pragma unroll
        for (int c = 0; c < 4; ++c) {
            const float ctr = -1.0f + (2.0f / 3.0f) * (float)c;
            bx[c] = fmaxf(0.0f, 1.0f - 1.5f * fabsf(u0 - ctr));
            by[c] = fmaxf(0.0f, 1.0f - 1.5f * fabsf(u1 - ctr));
        }

        // gather: this lane needs 8 contiguous source features (A layout),
        // converted to packed f16 once per tile (4 x v_cvt_pk_f16_f32)
        const float4* xrow = (const float4*)(x + (long)jn * 16 + half * 8);
        const float4 xlo = xrow[0];
        const float4 xhi = xrow[1];
        h2 xh[4];
        xh[0] = h2{(_Float16)xlo.x, (_Float16)xlo.y};
        xh[1] = h2{(_Float16)xlo.z, (_Float16)xlo.w};
        xh[2] = h2{(_Float16)xhi.x, (_Float16)xhi.y};
        xh[3] = h2{(_Float16)xhi.z, (_Float16)xhi.w};

        // --- GEMM over K=256 as 8 WMMA steps; step s covers basis terms 2s,2s+1 ---
        v8f acc = {};
#pragma unroll
        for (int s = 0; s < 8; ++s) {
            const int k0 = 2 * s, k1 = 2 * s + 1;
            const _Float16 b0 = (_Float16)(bx[k0 >> 2] * by[k0 & 3]);
            const _Float16 b1 = (_Float16)(bx[k1 >> 2] * by[k1 & 3]);
            const h2 b0p = h2{b0, b0};
            const h2 b1p = h2{b1, b1};
            AFrag A;
#pragma unroll
            for (int v = 0; v < 8; ++v) {
                // K = (v/4)*16 + half*8 + 2*(v%4) -> feature pair xh[v%4], basis (v/4)
                A.p[v] = (v >= 4 ? b1p : b0p) * xh[v & 3];   // v_pk_mul_f16
            }
            acc = __builtin_amdgcn_wmma_f32_16x16x32_f16(
                false, A.v, false, Bfrag[s], (short)0, acc, false, false);
        }

        // --- epilogue: C layout M = d + 8*half, N = ln ---
        // the 8 edges this lane scatters are contiguous: eBase + 8*half + 0..7
        const int4* pi = (const int4*)(ei + eBase + 8 * half);
        const int4* pj = (const int4*)(ej + eBase + 8 * half);
        const int4 i0 = pi[0], i1 = pi[1];
        const int4 j0 = pj[0], j1 = pj[1];
        const int iiv[8] = {i0.x, i0.y, i0.z, i0.w, i1.x, i1.y, i1.z, i1.w};
        const int jjv[8] = {j0.x, j0.y, j0.z, j0.w, j1.x, j1.y, j1.z, j1.w};
#pragma unroll
        for (int d = 0; d < 8; ++d) {
            if (iiv[d] != jjv[d]) {
                atomicAdd(&out[(long)iiv[d] * 16 + ln], acc[d] * OUT_SCALE);
            }
        }
    }
}

extern "C" void kernel_launch(void* const* d_in, const int* in_sizes, int n_in,
                              void* d_out, int out_size, void* d_ws, size_t ws_size,
                              hipStream_t stream) {
    const float* x  = (const float*)d_in[0];
    const float* ea = (const float*)d_in[1];
    const float* W  = (const float*)d_in[2];
    const int*   ei = (const int*)d_in[3];
    const int*   ej = (const int*)d_in[4];
    float* out = (float*)d_out;

    // out must start at zero for the atomic scatter-add (harness poisons it)
    zero_f32_kernel<<<(out_size + 255) / 256, 256, 0, stream>>>(out, out_size);

    // 1024 blocks x 8 waves = 8192 waves; ~6 tiles per wave amortizes the
    // one-time B-fragment preload over several WMMA tiles while keeping
    // enough waves in flight to hide gather/atomic latency.
    basis_conv_kernel<<<1024, 256, 0, stream>>>(x, ea, W, ei, ej, out);
}
